// sLSTMCell_38199439131020
// MI455X (gfx1250) — compile-verified, matrix operand order checked by reading, not run on hardware
//
#include <hip/hip_runtime.h>

typedef __attribute__((ext_vector_type(16))) __bf16 v16bf;
typedef __attribute__((ext_vector_type(8)))  float  v8f;
typedef __attribute__((ext_vector_type(4)))  float  v4f;   // builtin vector: AS-safe loads
typedef __attribute__((ext_vector_type(4)))  unsigned v4u;
typedef __attribute__((ext_vector_type(8)))  int      v8i;
typedef __attribute__((ext_vector_type(4)))  int      v4i;

// Address-space(1) pointers force global_load/global_store (not flat_*).
typedef const float __attribute__((address_space(1)))* gcf_p;
typedef const v4f   __attribute__((address_space(1)))* gcv4_p;
typedef float       __attribute__((address_space(1)))* gf_p;

#define H_DIM 2048
#define I_DIM 2048
#define B_DIM 4096
#define NT    128   // 2 phases (W@x, R@h) * 64 k-tiles of 32

// Block tile: 32 (M over H) x 128 (N over B), K-step 32.
// 256 threads = 8 waves: wave grid 2(M) x 4(N); per wave: 16x32 out sub-tile
// per gate => 4 gates x 2 accumulators (v8f) = 64 acc VGPRs / lane.
// A tiles (double buffered fp32 in LDS):
//   gates 0/1: TDM tensor_load_to_lds (TENSORcnt), one D# per gate, waves 0/1 issue
//   gates 2/3: per-lane global_load_async_to_lds_b128 (ASYNCcnt)
// B tiles: fp32->bf16 at stage time, stored [N][K] so fragments load directly.
__global__ __launch_bounds__(256)
void slstm_wmma_kernel(const float* __restrict__ x,
                       const float* __restrict__ h_prev,
                       const float* __restrict__ c_prev,
                       const float* __restrict__ n_prev,
                       const float* __restrict__ m_prev,
                       const float* __restrict__ w_i, const float* __restrict__ w_f,
                       const float* __restrict__ w_o, const float* __restrict__ w_z,
                       const float* __restrict__ r_i, const float* __restrict__ r_f,
                       const float* __restrict__ r_o, const float* __restrict__ r_z,
                       const float* __restrict__ b_i, const float* __restrict__ b_f,
                       const float* __restrict__ b_o, const float* __restrict__ b_z,
                       float* __restrict__ out)
{
    // Double-buffered LDS: A fp32 2*4*(32x32)*4B = 32 KB, B bf16 2*(128x32)*2B = 16 KB.
    __shared__ __align__(16) float  lds_a32[2][4][32 * 32];
    __shared__ __align__(32) __bf16 lds_bh [2][128 * 32];

    const int tid  = threadIdx.x;
    const int lane = tid & 31;
    const int wv   = tid >> 5;
    const int wm   = wv & 1;      // 2 waves along M -> rows wm*16 .. +15
    const int wn   = wv >> 1;     // 4 waves along N -> cols wn*32 .. +31
    const int lh   = lane & 15;
    const int kh   = lane >> 4;

    // Wave id as a compiler-visible SGPR value -> uniform scalar branch for
    // the TDM issue path.
    const int wvu = __builtin_amdgcn_readfirstlane(wv);

    const int m0 = blockIdx.x * 32;    // over H (gridDim.x = 64)
    const int n0 = blockIdx.y * 128;   // over B (gridDim.y = 32)

    // Staging coordinates.
    const int ar  = tid >> 3;          // A row 0..31 (per gate, 4 floats each)
    const int ak  = (tid & 7) << 2;    // A k offset 0,4,..,28 (16 B async chunks)
    const int bkr = tid >> 3;          // B k row 0..31
    const int bnc = (tid & 7) << 4;    // B n offset 0,16,..,112 (16 floats each)

    v8f acc[4][2];
#pragma unroll
    for (int g = 0; g < 4; ++g)
#pragma unroll
        for (int i = 0; i < 2; ++i)
            acc[g][i] = (v8f){0.f, 0.f, 0.f, 0.f, 0.f, 0.f, 0.f, 0.f};

    // ---- stage(ktn, buf): TDM + async-copy A tiles, cvt-stage B tile ----
    auto stage = [&](int ktn, int buf) {
        const int p  = ktn >> 6;           // 0: W@x, 1: R@h
        const int k0 = (ktn & 63) << 5;

        // Gates 0/1 via Tensor Data Mover: 2D tile 32(K) x 32(M) fp32.
        // D# group0: count=1 | lds_addr | global_addr | type=2.
        // D# group1: data_size=4B | tensor_dim0=2048 | tensor_dim1=2048 |
        //            tile_dim0=32 | tile_dim1=32 | tensor_dim0_stride=2048.
        if (wvu < 2) {
            const int g = wvu;
            const float* ab = (p == 0) ? ((g == 0) ? w_i : w_f)
                                       : ((g == 0) ? r_i : r_f);
            unsigned long long ga =
                (unsigned long long)(const void*)(ab + (size_t)m0 * I_DIM + k0);
            unsigned ldsa =
                (unsigned)(unsigned long long)(size_t)&lds_a32[buf][g][0];
            v4u d0 = { 1u,                               // count=1, user mode
                       ldsa,                             // lds_addr
                       (unsigned)ga,                     // global_addr[31:0]
                       (unsigned)(ga >> 32) | 0x80000000u }; // addr[56:32] | type=2
            v8i d1 = { 0x00020000,                       // data_size=2 (4 bytes)
                       0x08000000,                       // tensor_dim0[15:0]=2048 <<16
                       0x08000000,                       // tensor_dim1[15:0]=2048 <<16
                       0x00200000,                       // tile_dim0=32 <<16
                       32,                               // tile_dim1=32 (tile_dim2=0)
                       2048,                             // tensor_dim0_stride lo32
                       0, 0 };
            v4i z4 = { 0, 0, 0, 0 };
            v8i z8 = { 0, 0, 0, 0, 0, 0, 0, 0 };
            __builtin_amdgcn_tensor_load_to_lds(d0, d1, z4, z4, z8, 0);
        }

        // Gates 2/3 via per-lane async copy (16 B per thread per gate).
#pragma unroll
        for (int g = 2; g < 4; ++g) {
            const float* ab = (p == 0) ? ((g == 2) ? w_o : w_z)
                                       : ((g == 2) ? r_o : r_z);
            unsigned long long src =
                (unsigned long long)(const void*)(ab + (size_t)(m0 + ar) * I_DIM + (k0 + ak));
            unsigned lds_off =
                (unsigned)(unsigned long long)(size_t)&lds_a32[buf][g][ar * 32 + ak];
            asm volatile("global_load_async_to_lds_b128 %0, %1, off"
                         :: "v"(lds_off), "v"(src) : "memory");
        }

        // B tile: global_load (AS1) -> bf16 -> LDS transposed [N][K].
        const float* bb = (p == 0) ? x : h_prev;
        gcv4_p src = (gcv4_p)(const void*)(bb + (size_t)(k0 + bkr) * B_DIM + (n0 + bnc));
        v4f f0 = src[0];
        v4f f1 = src[1];
        v4f f2 = src[2];
        v4f f3 = src[3];
        __bf16* d = &lds_bh[buf][0];
        const int base = bnc * 32 + bkr;
        d[base +  0 * 32] = (__bf16)f0.x; d[base +  1 * 32] = (__bf16)f0.y;
        d[base +  2 * 32] = (__bf16)f0.z; d[base +  3 * 32] = (__bf16)f0.w;
        d[base +  4 * 32] = (__bf16)f1.x; d[base +  5 * 32] = (__bf16)f1.y;
        d[base +  6 * 32] = (__bf16)f1.z; d[base +  7 * 32] = (__bf16)f1.w;
        d[base +  8 * 32] = (__bf16)f2.x; d[base +  9 * 32] = (__bf16)f2.y;
        d[base + 10 * 32] = (__bf16)f2.z; d[base + 11 * 32] = (__bf16)f2.w;
        d[base + 12 * 32] = (__bf16)f3.x; d[base + 13 * 32] = (__bf16)f3.y;
        d[base + 14 * 32] = (__bf16)f3.z; d[base + 15 * 32] = (__bf16)f3.w;
    };

    // ---- compute(buf): 8 WMMAs on the current buffers ----
    auto compute = [&](int buf) {
        v16bf bfrag[2];
#pragma unroll
        for (int i = 0; i < 2; ++i)
            bfrag[i] = *(const v16bf*)&lds_bh[buf][(wn * 32 + i * 16 + lh) * 32 + kh * 16];
#pragma unroll
        for (int g = 0; g < 4; ++g) {
            const float* af = &lds_a32[buf][g][(wm * 16 + lh) * 32 + kh * 16];
            v4f a0 = *(const v4f*)(af + 0);
            v4f a1 = *(const v4f*)(af + 4);
            v4f a2 = *(const v4f*)(af + 8);
            v4f a3 = *(const v4f*)(af + 12);
            v16bf afrag;
            afrag[ 0] = (__bf16)a0.x; afrag[ 1] = (__bf16)a0.y;
            afrag[ 2] = (__bf16)a0.z; afrag[ 3] = (__bf16)a0.w;
            afrag[ 4] = (__bf16)a1.x; afrag[ 5] = (__bf16)a1.y;
            afrag[ 6] = (__bf16)a1.z; afrag[ 7] = (__bf16)a1.w;
            afrag[ 8] = (__bf16)a2.x; afrag[ 9] = (__bf16)a2.y;
            afrag[10] = (__bf16)a2.z; afrag[11] = (__bf16)a2.w;
            afrag[12] = (__bf16)a3.x; afrag[13] = (__bf16)a3.y;
            afrag[14] = (__bf16)a3.z; afrag[15] = (__bf16)a3.w;
#pragma unroll
            for (int i = 0; i < 2; ++i)
                acc[g][i] = __builtin_amdgcn_wmma_f32_16x16x32_bf16(
                    false, afrag, false, bfrag[i],
                    (short)0, acc[g][i], false, false);
        }
    };

    auto drain = [&]() {
        // Unconditional: waiting TENSORcnt<=0 is free on waves that issued no
        // TDM ops (counter already 0); avoids a per-iteration branch + spill.
        asm volatile("s_wait_asynccnt 0x0\n\t"
                     "s_wait_tensorcnt 0x0" ::: "memory");
        __syncthreads();   // everyone's LDS writes visible
    };

    // ---- software pipeline: issue k+1, compute k, wait, barrier ----
    stage(0, 0);
    drain();

#pragma unroll 1
    for (int kt = 0; kt < NT; kt += 2) {      // unroll-by-2: buffer ids are constants
        if (kt + 1 < NT) stage(kt + 1, 1);    // overlaps with compute(0)
        compute(0);
        drain();
        if (kt + 2 < NT) stage(kt + 2, 0);    // overlaps with compute(1)
        compute(1);
        drain();
    }

    // ---- Epilogue: stabilized exponential gating (fp32, global_* via AS1) ----
    const size_t HB = (size_t)H_DIM * B_DIM;
    gcf_p cp  = (gcf_p)(const void*)c_prev;
    gcf_p np  = (gcf_p)(const void*)n_prev;
    gcf_p mp_ = (gcf_p)(const void*)m_prev;
    gcf_p bi  = (gcf_p)(const void*)b_i;
    gcf_p bf  = (gcf_p)(const void*)b_f;
    gcf_p bo  = (gcf_p)(const void*)b_o;
    gcf_p bz  = (gcf_p)(const void*)b_z;
    gf_p  og  = (gf_p)(void*)out;

#pragma unroll
    for (int i = 0; i < 2; ++i) {
        const int col = n0 + wn * 32 + i * 16 + lh;
#pragma unroll
        for (int r = 0; r < 8; ++r) {
            const int row = m0 + wm * 16 + r + kh * 8;   // C/D layout: M = r + 8*(lane>>4)
            const size_t off = (size_t)row * B_DIM + col;

            const float iw = acc[0][i][r] + bi[row];
            const float fw = acc[1][i][r] + bf[row];
            const float ow = acc[2][i][r] + bo[row];
            const float zw = acc[3][i][r] + bz[row];

            const float log_i = iw;
            const float log_f = fminf(fw, 0.f) - log1pf(__expf(-fabsf(fw)));
            const float mv    = mp_[off];
            const float m_t   = fmaxf(log_f + mv, log_i);
            const float ip    = __expf(log_i - m_t);
            const float fp    = __expf(log_f + mv - m_t);
            const float c_t   = fp * cp[off] + ip * tanhf(zw);
            const float n_t   = fp * np[off] + fp;
            const float h_t   = (1.f / (1.f + __expf(-ow))) * tanhf(c_t / n_t);

            og[off]          = h_t;   // h_t (first tuple element)
            og[HB + off]     = h_t;   // h_t (state tuple)
            og[2 * HB + off] = c_t;
            og[3 * HB + off] = n_t;
            og[4 * HB + off] = m_t;
        }
    }
}

extern "C" void kernel_launch(void* const* d_in, const int* in_sizes, int n_in,
                              void* d_out, int out_size, void* d_ws, size_t ws_size,
                              hipStream_t stream) {
    const float* x      = (const float*)d_in[0];
    const float* h_prev = (const float*)d_in[1];
    const float* c_prev = (const float*)d_in[2];
    const float* n_prev = (const float*)d_in[3];
    const float* m_prev = (const float*)d_in[4];
    const float* w_i    = (const float*)d_in[5];
    const float* w_f    = (const float*)d_in[6];
    const float* w_o    = (const float*)d_in[7];
    const float* w_z    = (const float*)d_in[8];
    const float* r_i    = (const float*)d_in[9];
    const float* r_f    = (const float*)d_in[10];
    const float* r_o    = (const float*)d_in[11];
    const float* r_z    = (const float*)d_in[12];
    const float* b_i    = (const float*)d_in[13];
    const float* b_f    = (const float*)d_in[14];
    const float* b_o    = (const float*)d_in[15];
    const float* b_z    = (const float*)d_in[16];
    float* out = (float*)d_out;

    dim3 grid(H_DIM / 32, B_DIM / 128, 1);  // 64 x 32 blocks
    slstm_wmma_kernel<<<grid, 256, 0, stream>>>(
        x, h_prev, c_prev, n_prev, m_prev,
        w_i, w_f, w_o, w_z, r_i, r_f, r_o, r_z,
        b_i, b_f, b_o, b_z, out);
}
